// GPRGNN_78365973283180
// MI455X (gfx1250) — matrix operand order repaired and checked by reading, not compiled, not run on hardware
//
#include <hip/hip_runtime.h>
#include <hip/hip_bf16.h>

typedef __attribute__((ext_vector_type(16))) _Float16 v16h;
typedef __attribute__((ext_vector_type(8)))  _Float16 v8h;
typedef __attribute__((ext_vector_type(8)))  float    v8f;

#define F_DIM 256
#define N_CLASSES 64
#define BM 64
#define BN 64
#define BK 32

// ---------------------------------------------------------------------------
// Degree / norm kernels
// ---------------------------------------------------------------------------
__global__ void gpr_init_deg(float* deg, int n) {
    int i = blockIdx.x * blockDim.x + threadIdx.x;
    if (i < n) deg[i] = 1.0f;  // self-loop contributes 1
}

__global__ void gpr_count_deg(const long long* col, float* deg, long long E) {
    long long e = (long long)blockIdx.x * blockDim.x + threadIdx.x;
    if (e < E) atomicAdd(&deg[(int)col[e]], 1.0f);
}

__global__ void gpr_make_dinv(float* deg, int n) {
    int i = blockIdx.x * blockDim.x + threadIdx.x;
    if (i < n) {
        float d = deg[i];
        deg[i] = (d > 0.0f) ? rsqrtf(d) : 0.0f;
    }
}

// ---------------------------------------------------------------------------
// WMMA GEMM: out[M,N] = act(A[M,K] @ B[K,N] + bias)
// Block tile 64x64, 8 waves (wave32); each wave computes a 16x32 strip
// (2 WMMA tiles sharing the A fragment). LDS layouts chosen so fragment
// loads are contiguous 128-bit DS reads with k = 16*lane_half + s in both A and B.
// ---------------------------------------------------------------------------
template <typename AT, typename OT, bool PRELU>
__global__ __launch_bounds__(256)
void gpr_wmma_gemm(const AT* __restrict__ A, const float* __restrict__ B,
                   const float* __restrict__ bias, const float* __restrict__ prelu_w,
                   OT* __restrict__ out, int M, int N, int K) {
    __shared__ _Float16 sA[BM * BK];   // row-major [m][k], 4 KB
    __shared__ _Float16 sBt[BN * BK];  // transposed  [n][k], 4 KB

    const int tid    = threadIdx.x;
    const int lane   = tid & 31;
    const int warpId = tid >> 5;        // 0..7
    const int warp_m = warpId & 3;      // 4 strips of 16 rows
    const int warp_n = warpId >> 2;     // 2 strips of 32 cols
    const int lg     = lane >> 4;       // lane half (0/1)
    const int l15    = lane & 15;

    const int block_m = blockIdx.y * BM;
    const int block_n = blockIdx.x * BN;

    // Staging coordinates (uniform per thread, hoisted out of K loop)
    const int a_row  = tid >> 2;          // 0..63
    const int a_kseg = (tid & 3) << 3;    // 0,8,16,24
    const int b_n    = tid & 63;          // 0..63
    const int b_kseg = (tid >> 6) << 3;   // 0,8,16,24

    float pw = 0.0f;
    if constexpr (PRELU) pw = prelu_w[0];

    v8f acc0 = {0.f, 0.f, 0.f, 0.f, 0.f, 0.f, 0.f, 0.f};
    v8f acc1 = {0.f, 0.f, 0.f, 0.f, 0.f, 0.f, 0.f, 0.f};

    for (int k0 = 0; k0 < K; k0 += BK) {
        // ---- Stage A tile: one vector load + one ds_store_b128 per thread
        {
            v8h av = {};
            const int gm = block_m + a_row;
            if (gm < M) {
                if constexpr (sizeof(AT) == 4) {
                    const float4 f0 = *(const float4*)((const float*)A + (size_t)gm * K + k0 + a_kseg);
                    const float4 f1 = *(const float4*)((const float*)A + (size_t)gm * K + k0 + a_kseg + 4);
                    av[0] = (_Float16)f0.x; av[1] = (_Float16)f0.y;
                    av[2] = (_Float16)f0.z; av[3] = (_Float16)f0.w;
                    av[4] = (_Float16)f1.x; av[5] = (_Float16)f1.y;
                    av[6] = (_Float16)f1.z; av[7] = (_Float16)f1.w;
                } else {
                    av = *(const v8h*)((const _Float16*)A + (size_t)gm * K + k0 + a_kseg);
                }
            }
            *(v8h*)(&sA[a_row * BK + a_kseg]) = av;
        }
        // ---- Stage B tile transposed: 8 coalesced dword loads, one b128 store
        {
            v8h bv;
            #pragma unroll
            for (int j = 0; j < 8; ++j)
                bv[j] = (_Float16)B[(size_t)(k0 + b_kseg + j) * N + block_n + b_n];
            *(v8h*)(&sBt[b_n * BK + b_kseg]) = bv;
        }
        __syncthreads();

        // ---- A fragment: two contiguous 16B runs (K base lg*16, +8)
        const int arow = warp_m * 16 + l15;
        const v8h a0 = *(const v8h*)(&sA[arow * BK + (lg << 4)]);
        const v8h a1 = *(const v8h*)(&sA[arow * BK + 8 + (lg << 4)]);
        const v16h af = __builtin_shufflevector(a0, a1,
            0, 1, 2, 3, 4, 5, 6, 7, 8, 9, 10, 11, 12, 13, 14, 15);

        // ---- B fragments: one contiguous 32B run each (K base lg*16)
        const int nbase = warp_n * 32 + l15;
        const v16h bf0 = *(const v16h*)(&sBt[nbase * BK + (lg << 4)]);
        const v16h bf1 = *(const v16h*)(&sBt[(nbase + 16) * BK + (lg << 4)]);

        acc0 = __builtin_amdgcn_wmma_f32_16x16x32_f16(
            false, af, false, bf0, (short)0, acc0, false, false);
        acc1 = __builtin_amdgcn_wmma_f32_16x16x32_f16(
            false, af, false, bf1, (short)0, acc1, false, false);

        __syncthreads();
    }

    // Epilogue: C/D layout — VGPR r: lanes 0-15 hold M=r, lanes 16-31 hold M=r+8
    const int ncol0 = block_n + warp_n * 32 + l15;
    const int ncol1 = ncol0 + 16;
    const float bias0 = bias[ncol0];
    const float bias1 = bias[ncol1];
    #pragma unroll
    for (int r = 0; r < 8; ++r) {
        int m = block_m + warp_m * 16 + (lg ? r + 8 : r);
        if (m < M) {
            float v0 = acc0[r] + bias0;
            float v1 = acc1[r] + bias1;
            if constexpr (PRELU) {
                v0 = (v0 >= 0.0f) ? v0 : pw * v0;
                v1 = (v1 >= 0.0f) ? v1 : pw * v1;
            }
            out[(size_t)m * N + ncol0] = (OT)v0;
            out[(size_t)m * N + ncol1] = (OT)v1;
        }
    }
}

// ---------------------------------------------------------------------------
// Propagation kernels (feature dim = 64, float4 vectorized => 16 float4/node)
// ---------------------------------------------------------------------------
__global__ void gpr_init_hidden(const float* __restrict__ h, float* __restrict__ hidden,
                                const float* __restrict__ temp, int n16) {
    int i = blockIdx.x * blockDim.x + threadIdx.x;
    if (i >= n16) return;
    float t0 = temp[0];
    float4 v = ((const float4*)h)[i];
    ((float4*)hidden)[i] = make_float4(t0 * v.x, t0 * v.y, t0 * v.z, t0 * v.w);
}

__global__ void gpr_self_init(const float* __restrict__ dinv, const float* __restrict__ src,
                              float* __restrict__ dst, int n16) {
    int i = blockIdx.x * blockDim.x + threadIdx.x;
    if (i >= n16) return;
    int node = i >> 4;
    float d = dinv[node];
    float w = d * d;
    float4 v = ((const float4*)src)[i];
    ((float4*)dst)[i] = make_float4(w * v.x, w * v.y, w * v.z, w * v.w);
}

__global__ void gpr_edge_scatter(const long long* __restrict__ row,
                                 const long long* __restrict__ col,
                                 const float* __restrict__ dinv,
                                 const float* __restrict__ src,
                                 float* __restrict__ dst, long long E) {
    long long t = (long long)blockIdx.x * blockDim.x + threadIdx.x;
    long long e = t >> 4;                 // 16 threads per edge (64 feats / float4)
    if (e >= E) return;
    int c = (int)(t & 15) * 4;
    int r  = (int)row[e];
    int cl = (int)col[e];
    float w = dinv[r] * dinv[cl];
    float4 v = *(const float4*)(src + (size_t)r * 64 + c);
    float* d = dst + (size_t)cl * 64 + c;
    atomicAdd(d + 0, w * v.x);
    atomicAdd(d + 1, w * v.y);
    atomicAdd(d + 2, w * v.z);
    atomicAdd(d + 3, w * v.w);
}

__global__ void gpr_axpy_hidden(float* __restrict__ hidden, const float* __restrict__ src,
                                const float* __restrict__ temp, int k, int n16) {
    int i = blockIdx.x * blockDim.x + threadIdx.x;
    if (i >= n16) return;
    float g = temp[k];
    float4 h = ((float4*)hidden)[i];
    float4 v = ((const float4*)src)[i];
    h.x += g * v.x; h.y += g * v.y; h.z += g * v.z; h.w += g * v.w;
    ((float4*)hidden)[i] = h;
}

// ---------------------------------------------------------------------------
// Row-wise log-softmax over 64 classes: 1 wave32 per row, 2 classes per lane
// ---------------------------------------------------------------------------
__global__ void gpr_log_softmax(const float* __restrict__ hidden, float* __restrict__ out, int n) {
    int rowsPerBlock = blockDim.x >> 5;
    int row = blockIdx.x * rowsPerBlock + (threadIdx.x >> 5);
    int lane = threadIdx.x & 31;
    if (row >= n) return;
    const float* hrow = hidden + (size_t)row * 64;
    float v0 = hrow[lane];
    float v1 = hrow[lane + 32];
    float m = fmaxf(v0, v1);
    #pragma unroll
    for (int off = 16; off > 0; off >>= 1)
        m = fmaxf(m, __shfl_xor(m, off, 32));
    float s = __expf(v0 - m) + __expf(v1 - m);
    #pragma unroll
    for (int off = 16; off > 0; off >>= 1)
        s += __shfl_xor(s, off, 32);
    float l = __logf(s);
    float* orow = out + (size_t)row * 64;
    orow[lane]      = v0 - m - l;
    orow[lane + 32] = v1 - m - l;
}

// ---------------------------------------------------------------------------
// Host orchestration
// ---------------------------------------------------------------------------
static inline size_t align_up(size_t x) { return (x + 511) & ~(size_t)511; }

extern "C" void kernel_launch(void* const* d_in, const int* in_sizes, int n_in,
                              void* d_out, int out_size, void* d_ws, size_t ws_size,
                              hipStream_t stream) {
    const float*     x       = (const float*)d_in[0];
    const long long* ei      = (const long long*)d_in[1];
    const float*     W1      = (const float*)d_in[2];
    const float*     b1      = (const float*)d_in[3];
    const float*     prelu_w = (const float*)d_in[4];
    const float*     W2      = (const float*)d_in[5];
    const float*     b2      = (const float*)d_in[6];
    const float*     temp    = (const float*)d_in[7];

    const int n  = in_sizes[0] / F_DIM;          // 100000
    const long long E = (long long)in_sizes[1] / 2;
    const int C  = in_sizes[6];                  // 64
    const int Kh = in_sizes[7] - 1;              // 10

    const long long* erow = ei;
    const long long* ecol = ei + E;

    // Workspace layout
    char* w = (char*)d_ws;
    size_t off = 0;
    float* dinv = (float*)(w + off);      off += align_up((size_t)n * 4);
    _Float16* hh = (_Float16*)(w + off);  off += align_up((size_t)n * F_DIM * 2);
    float* p0 = (float*)(w + off);        off += align_up((size_t)n * C * 4);
    float* p1 = (float*)(w + off);        off += align_up((size_t)n * C * 4);
    float* hid = (float*)(w + off);       off += align_up((size_t)n * C * 4);
    (void)ws_size; (void)n_in; (void)out_size;

    const int T = 256;
    const int n16 = n * (C / 4);  // float4 count for [n, 64]

    // 1) degrees -> dinv
    gpr_init_deg<<<(n + T - 1) / T, T, 0, stream>>>(dinv, n);
    gpr_count_deg<<<(int)((E + T - 1) / T), T, 0, stream>>>(ecol, dinv, E);
    gpr_make_dinv<<<(n + T - 1) / T, T, 0, stream>>>(dinv, n);

    // 2) GEMM1: h1 = PReLU(x @ W1 + b1)  -> f16
    {
        dim3 grid(F_DIM / BN, (n + BM - 1) / BM);
        gpr_wmma_gemm<float, _Float16, true><<<grid, T, 0, stream>>>(
            x, W1, b1, prelu_w, hh, n, F_DIM, F_DIM);
    }
    // 3) GEMM2: h = h1 @ W2 + b2 -> p0 (f32)
    {
        dim3 grid(C / BN, (n + BM - 1) / BM);
        gpr_wmma_gemm<_Float16, float, false><<<grid, T, 0, stream>>>(
            hh, W2, b2, nullptr, p0, n, C, F_DIM);
    }

    // 4) hidden = temp[0] * h ; hk = p0
    gpr_init_hidden<<<(n16 + T - 1) / T, T, 0, stream>>>(p0, hid, temp, n16);

    // 5) K-hop propagation with ping-pong buffers
    float* src = p0;
    float* dst = p1;
    for (int k = 1; k <= Kh; ++k) {
        gpr_self_init<<<(n16 + T - 1) / T, T, 0, stream>>>(dinv, src, dst, n16);
        long long nt = E * 16;
        gpr_edge_scatter<<<(int)((nt + T - 1) / T), T, 0, stream>>>(
            erow, ecol, dinv, src, dst, E);
        gpr_axpy_hidden<<<(n16 + T - 1) / T, T, 0, stream>>>(hid, dst, temp, k, n16);
        float* tmp = src; src = dst; dst = tmp;
    }

    // 6) log-softmax -> d_out
    {
        int rowsPerBlock = T / 32;
        gpr_log_softmax<<<(n + rowsPerBlock - 1) / rowsPerBlock, T, 0, stream>>>(
            hid, (float*)d_out, n);
    }
}
